// InteractionLayer_27659589386344
// MI455X (gfx1250) — matrix operand order repaired and checked by reading, not compiled, not run on hardware
//
#include <hip/hip_runtime.h>

// ---------------------------------------------------------------------------
// Problem constants (match reference)
// ---------------------------------------------------------------------------
#define DM 1024   // D_MANIFOLD
#define DF 512    // D_FIBER
#define SQn 4096  // SQ
#define SKn 4096  // SK
#define NB 4      // batch

typedef __attribute__((ext_vector_type(16))) __bf16 v16bf;
typedef __attribute__((ext_vector_type(8)))  float  v8f;

union BFrag { v16bf v; unsigned u[8]; };

// Probe for the Tensor Data Mover builtin (arity differs across toolchains)
#if defined(__has_builtin)
#  if __has_builtin(__builtin_amdgcn_tensor_load_to_lds) && \
      __has_builtin(__builtin_amdgcn_s_wait_tensorcnt)
#    define USE_TDM 1
#  else
#    define USE_TDM 0
#  endif
#else
#  define USE_TDM 0
#endif

typedef unsigned __attribute__((ext_vector_type(4))) u32x4;
typedef int      __attribute__((ext_vector_type(4))) i32x4;
typedef int      __attribute__((ext_vector_type(8))) i32x8;
typedef __attribute__((address_space(3))) unsigned short lds_us_t;

// f32 -> bf16 round-to-nearest-even (bit twiddle; no __bf16 scalar math needed)
__device__ __forceinline__ unsigned short f2bf(float f) {
    unsigned u = __float_as_uint(f);
    u += 0x7FFFu + ((u >> 16) & 1u);
    return (unsigned short)(u >> 16);
}

// ---------------------------------------------------------------------------
// Kernel 0: W[k][n] f32 -> Wt[n][k] bf16 (transposed so B-fragment K-pairs
// are contiguous in memory / LDS)
// ---------------------------------------------------------------------------
__global__ void wt_convert(const float* __restrict__ W,
                           unsigned short* __restrict__ Wt, int Kdim) {
    int idx = blockIdx.x * blockDim.x + threadIdx.x;
    if (idx >= Kdim * DF) return;
    int k = idx / DF, n = idx % DF;
    Wt[(size_t)n * Kdim + k] = f2bf(W[idx]);
}

// ---------------------------------------------------------------------------
// Kernel 1: projection GEMM  Y = bf16( X[rows][Kdim] * W[Kdim][DF] + b )
// Block tile 128(M) x 64(N), 8 waves, each wave 16x64 (4 WMMA N-subtiles).
// rows = 16384 (divisible by 128), grid = (rows/128, DF/64).
// ---------------------------------------------------------------------------
#define A_LD 40   // 128x32 bf16 A tile, padded (80B row stride -> no conflicts)
#define B_LD 40   // 64x32 bf16 B tile (stored [n][k])

__global__ void __launch_bounds__(256)
proj_gemm(const float* __restrict__ X, const unsigned short* __restrict__ Wt,
          const float* __restrict__ bias, unsigned short* __restrict__ Y,
          int Kdim) {
    __shared__ unsigned short Al[128 * A_LD];
    __shared__ unsigned short Bl[64 * B_LD];

    const int tid  = threadIdx.x;
    const int wave = tid >> 5;
    const int lane = tid & 31;
    const int nlan = lane & 15;
    const int half = lane >> 4;
    const size_t M0 = (size_t)blockIdx.x * 128;
    const int N0 = blockIdx.y * 64;

    v8f acc[4] = {};

    for (int k0 = 0; k0 < Kdim; k0 += 32) {
        __syncthreads();  // previous iteration fragment reads complete
        // Stage A: 128x32, f32 -> bf16, coalesced (32 lanes cover one row)
        #pragma unroll
        for (int i = 0; i < 16; ++i) {
            int idx = i * 256 + tid;
            int r = idx >> 5, k = idx & 31;
            Al[r * A_LD + k] = f2bf(X[(M0 + r) * Kdim + k0 + k]);
        }
        // Stage B: 64x32 from Wt[n][k] (already bf16, contiguous along k)
        #pragma unroll
        for (int i = 0; i < 8; ++i) {
            int idx = i * 256 + tid;
            int n = idx >> 5, k = idx & 31;
            Bl[n * B_LD + k] = Wt[(size_t)(N0 + n) * Kdim + k0 + k];
        }
        // Prefetch next A k-chunk (one 128B line covers the 32 f32 row chunk)
        if (k0 + 32 < Kdim) {
            int r = tid & 127;
            __builtin_prefetch(&X[(M0 + r) * Kdim + k0 + 32], 0, 1);
        }
        __syncthreads();

        // A fragment: 16x32 bf16, ISA layout: lane(m, half), VGPR j holds
        // K pair = half*8+2j (j<4) or 16+half*8+2(j-4)
        BFrag a;
        {
            int m = wave * 16 + nlan;
            #pragma unroll
            for (int j = 0; j < 8; ++j) {
                int kk = (j < 4) ? (half * 8 + 2 * j)
                                 : (16 + half * 8 + 2 * (j - 4));
                a.u[j] = *(const unsigned*)&Al[m * A_LD + kk];
            }
        }
        #pragma unroll
        for (int s = 0; s < 4; ++s) {
            // B fragment 32x16: same K order as A (half*8+2j / 16+half*8+2(j-4))
            BFrag b;
            int n = s * 16 + nlan;
            #pragma unroll
            for (int j = 0; j < 8; ++j) {
                int kk = (j < 4) ? (half * 8 + 2 * j) : (16 + half * 8 + 2 * (j - 4));
                b.u[j] = *(const unsigned*)&Bl[n * B_LD + kk];
            }
            acc[s] = __builtin_amdgcn_wmma_f32_16x16x32_bf16(
                false, a.v, false, b.v, (short)0, acc[s], false, false);
        }
    }

    // Epilogue: bias add (f32) then bf16 store. C layout: VGPR r -> M=r+8*half
    #pragma unroll
    for (int s = 0; s < 4; ++s) {
        int col = N0 + s * 16 + nlan;
        float bv = bias[col];
        #pragma unroll
        for (int r = 0; r < 8; ++r) {
            int m = wave * 16 + r + 8 * half;
            Y[(M0 + m) * DF + col] = f2bf(acc[s][r] + bv);
        }
    }
}

// ---------------------------------------------------------------------------
// Kernel 2: fused flash attention over bf16 Q,K,V -> f32 out
// Block = 256 thr (8 waves) = one 32-row query tile of one batch.
// Wave w owns output d-slice [w*64, w*64+64) for BOTH 16-row query sub-tiles.
// Key tiles of Bc=32.  K tile staged by the Tensor Data Mover (padded-row
// DMA reproduces K_LD=516 in hardware); V^T staged via register 2x2
// transpose (b32 LDS stores).  K^T/V B-fragments are reused across the two
// query sub-tiles, so each staged K/V byte feeds 2x the WMMA work.
// ---------------------------------------------------------------------------
#define K_LD 516  // K tile row stride (elems): 1032B -> 2-bank rotation
#define V_LD 34   // V^T row stride: 68B -> 17-bank rotation (even for b32 st)
#define S_LD 33
#define P_LD 36

// Issue one TDM descriptor: copy [32 rows x 512 bf16] global tile into LDS
// with 4-element (8B) padding after each 512-element row -> K_LD = 516.
__device__ __forceinline__ void tdm_load_k_tile(const unsigned short* gsrc,
                                                unsigned lds_byte_addr) {
#if USE_TDM
    unsigned long long ga = (unsigned long long)(size_t)gsrc;
    u32x4 g0;
    g0[0] = 1u;                                   // count=1 valid descriptor
    g0[1] = lds_byte_addr;                        // lds_addr
    g0[2] = (unsigned)ga;                         // global_addr[31:0]
    g0[3] = (unsigned)((ga >> 32) & 0x1FFFFFFull) // global_addr[56:32]
            | (2u << 30);                         // type=2 ("image")
    i32x8 g1;
    g1[0] = (int)((1u << 16)      // data_size = 2 bytes
                | (1u << 20)      // pad_enable
                | (7u << 22)      // pad_interval = 256 dwords (one row)
                | (1u << 25));    // pad_amount = 2 dwords (8B = 4 elems)
    g1[1] = (int)((unsigned)(DF & 0xFFFF) << 16);     // tensor_dim0[15:0]
    g1[2] = (int)((DF >> 16) | (32u << 16));          // dim0[31:16]|dim1[15:0]
    g1[3] = (int)((unsigned)DF << 16);                // dim1[31:16]|tile_dim0
    g1[4] = 32;                                       // tile_dim1=32, dim2=0
    g1[5] = DF;                                       // tensor_dim0_stride
    g1[6] = 0;
    g1[7] = 0;
    i32x4 z4 = (i32x4)0;
#if defined(__clang_major__) && __clang_major__ >= 23
    i32x8 z8 = (i32x8)0;
    __builtin_amdgcn_tensor_load_to_lds(g0, g1, z4, z4, z8, 0);
#else
    __builtin_amdgcn_tensor_load_to_lds(g0, g1, z4, z4, 0);
#endif
#else
    (void)gsrc; (void)lds_byte_addr;
#endif
}

__global__ void __launch_bounds__(256)
attn_fused(const unsigned short* __restrict__ Qb,
           const unsigned short* __restrict__ Kb,
           const unsigned short* __restrict__ Vb,
           float* __restrict__ Out) {
    __shared__ unsigned short KV[512 * V_LD];  // 17408 elems; K phase: 32*K_LD=16512
    __shared__ float Sl[32 * S_LD];            // 32x32 scores (padded)
    __shared__ unsigned short Pl[32 * P_LD];   // 32x32 bf16 probabilities
    __shared__ float mrow[32], lrow[32], arow[32];

    const int tid  = threadIdx.x;
    const int wave = tid >> 5;
    const int lane = tid & 31;
    const int nlan = lane & 15;
    const int half = lane >> 4;
    const int q0 = blockIdx.x * 32;
    const int bt = blockIdx.y;
    const int dbase = wave * 64;

    if (tid < 32) { mrow[tid] = -1e30f; lrow[tid] = 0.f; }

#if USE_TDM
    const unsigned kv_lds = (unsigned)(size_t)(lds_us_t*)KV;
#endif

    // Q fragments: 2 query sub-tiles x 2 d-chunks of this wave's 64-d slice
    BFrag aq[2][2];
    #pragma unroll
    for (int t = 0; t < 2; ++t) {
        size_t qrow = ((size_t)bt * SQn + q0 + t * 16 + nlan) * DF;
        #pragma unroll
        for (int c = 0; c < 2; ++c)
            #pragma unroll
            for (int j = 0; j < 8; ++j) {
                int kk = (j < 4) ? (half * 8 + 2 * j)
                                 : (16 + half * 8 + 2 * (j - 4));
                aq[t][c].u[j] = *(const unsigned*)&Qb[qrow + dbase + c * 32 + kk];
            }
    }

    v8f accO[2][4] = {};
    const float scale = 0.044194173824159216f;  // 1/sqrt(512)

    for (int kt = 0; kt < SKn / 32; ++kt) {
        __syncthreads();  // prior PV reads of KV done / stats init visible
        size_t base = ((size_t)bt * SKn + (size_t)kt * 32) * DF;

        // ---- stage K tile [32 keys][512 d] into KV (TDM DMA when available)
#if USE_TDM
        if (wave == 0) tdm_load_k_tile(&Kb[base], kv_lds);
#else
        #pragma unroll
        for (int i = 0; i < 32; ++i) {
            int dw = i * 256 + tid;        // 8192 dwords total
            int key = dw >> 8;             // 256 dwords per key row
            int dp = (dw & 255) * 2;
            unsigned u = *(const unsigned*)&Kb[base + (size_t)key * DF + dp];
            *(unsigned*)&KV[key * K_LD + dp] = u;
        }
#endif
        // ---- zero S accumulation tile (32*33 = 1056 floats)
        Sl[tid] = 0.f; Sl[256 + tid] = 0.f; Sl[512 + tid] = 0.f; Sl[768 + tid] = 0.f;
        if (tid < 32) Sl[1024 + tid] = 0.f;
        // ---- prefetch next K/V tiles (each thread one 128B line -> 32KB)
        if (kt + 1 < SKn / 32) {
            size_t nb = base + 32 * (size_t)DF;
            __builtin_prefetch(&Kb[nb + (size_t)tid * 64], 0, 1);
            __builtin_prefetch(&Vb[nb + (size_t)tid * 64], 0, 1);
        }
#if USE_TDM
        if (wave == 0) __builtin_amdgcn_s_wait_tensorcnt((short)0);
#endif
        __syncthreads();

        // ---- partial S = Q_slice * K_slice^T; K^T B-frag reused for both
        //      query sub-tiles
        v8f accS[2][2] = {};
        #pragma unroll
        for (int c = 0; c < 2; ++c) {
            #pragma unroll
            for (int nt = 0; nt < 2; ++nt) {
                BFrag b;                       // B = K^T: col n = key, K = d
                int key = nt * 16 + nlan;
                #pragma unroll
                for (int j = 0; j < 8; ++j) {
                    int kk = dbase + c * 32 + ((j < 4) ? (half * 8 + 2 * j) : (16 + half * 8 + 2 * (j - 4)));
                    b.u[j] = *(const unsigned*)&KV[key * K_LD + kk];
                }
                #pragma unroll
                for (int t = 0; t < 2; ++t)
                    accS[t][nt] = __builtin_amdgcn_wmma_f32_16x16x32_bf16(
                        false, aq[t][c].v, false, b.v, (short)0,
                        accS[t][nt], false, false);
            }
        }
        // ---- cross-wave d-reduction of S through LDS
        #pragma unroll
        for (int t = 0; t < 2; ++t)
            #pragma unroll
            for (int nt = 0; nt < 2; ++nt)
                #pragma unroll
                for (int r = 0; r < 8; ++r)
                    atomicAdd(&Sl[(t * 16 + r + 8 * half) * S_LD + nt * 16 + nlan],
                              accS[t][nt][r]);
        __syncthreads();

        // ---- stage V^T via register 2x2 transpose: two key rows' d-pair
        //      dwords repacked into two b32 stores at Vt[d][key], Vt[d+1][key]
        #pragma unroll
        for (int kp = 0; kp < 16; ++kp) {      // key pair index
            int d = (tid & 255) * 2;           // d pair base
            const unsigned short* v0 = &Vb[base + (size_t)(2 * kp) * DF + d];
            unsigned u0 = *(const unsigned*)v0;
            unsigned u1 = *(const unsigned*)(v0 + DF);
            *(unsigned*)&KV[(d)     * V_LD + 2 * kp] = (u0 & 0xFFFFu) | (u1 << 16);
            *(unsigned*)&KV[(d + 1) * V_LD + 2 * kp] = (u0 >> 16) | (u1 & 0xFFFF0000u);
        }
        // ---- parallel online softmax: two passes over the 32 rows; thread
        //      (row = p*16 + tid>>4, c = tid&15) handles keys c and c+16,
        //      shfl_xor reductions over the 16-lane row group
        #pragma unroll
        for (int p = 0; p < 2; ++p) {
            int r = p * 16 + (tid >> 4);
            int c = tid & 15;
            float s0 = Sl[r * S_LD + c] * scale;
            float s1 = Sl[r * S_LD + c + 16] * scale;
            float mold = mrow[r];
            float mx = fmaxf(s0, s1);
            #pragma unroll
            for (int m = 8; m >= 1; m >>= 1)
                mx = fmaxf(mx, __shfl_xor(mx, m, 16));
            mx = fmaxf(mx, mold);
            float p0 = __expf(s0 - mx);
            float p1 = __expf(s1 - mx);
            Pl[r * P_LD + c]      = f2bf(p0);
            Pl[r * P_LD + c + 16] = f2bf(p1);
            float sum = p0 + p1;
            #pragma unroll
            for (int m = 8; m >= 1; m >>= 1)
                sum += __shfl_xor(sum, m, 16);
            if (c == 0) {
                float al = __expf(mold - mx);
                arow[r] = al;
                mrow[r] = mx;
                lrow[r] = lrow[r] * al + sum;
            }
        }
        __syncthreads();

        // ---- PV: rescale O by alpha, accumulate P(32x32) * V(32 x 64slice);
        //      V B-frag built once per s and reused for both query sub-tiles
        float alr[2][8];
        #pragma unroll
        for (int t = 0; t < 2; ++t)
            #pragma unroll
            for (int r = 0; r < 8; ++r)
                alr[t][r] = arow[t * 16 + r + 8 * half];
        BFrag ap[2];
        #pragma unroll
        for (int t = 0; t < 2; ++t)
            #pragma unroll
            for (int j = 0; j < 8; ++j) {
                int kk = (j < 4) ? (half * 8 + 2 * j)
                                 : (16 + half * 8 + 2 * (j - 4));
                ap[t].u[j] = *(const unsigned*)&Pl[(t * 16 + nlan) * P_LD + kk];
            }
        #pragma unroll
        for (int s = 0; s < 4; ++s) {
            BFrag bv;                          // B = V: col n = d, K = key
            int dcol = dbase + s * 16 + nlan;
            #pragma unroll
            for (int j = 0; j < 8; ++j) {
                int kk = (j < 4) ? (half * 8 + 2 * j) : (16 + half * 8 + 2 * (j - 4));
                bv.u[j] = *(const unsigned*)&KV[dcol * V_LD + kk];
            }
            #pragma unroll
            for (int t = 0; t < 2; ++t) {
                #pragma unroll
                for (int r = 0; r < 8; ++r)
                    accO[t][s][r] *= alr[t][r];
                accO[t][s] = __builtin_amdgcn_wmma_f32_16x16x32_bf16(
                    false, ap[t].v, false, bv.v, (short)0, accO[t][s], false, false);
            }
        }
    }

    // ---- finalize: divide by running sum, store f32
    float lr[2][8];
    #pragma unroll
    for (int t = 0; t < 2; ++t)
        #pragma unroll
        for (int r = 0; r < 8; ++r)
            lr[t][r] = lrow[t * 16 + r + 8 * half];
    size_t outbase = ((size_t)bt * SQn + q0) * DF;
    #pragma unroll
    for (int t = 0; t < 2; ++t)
        #pragma unroll
        for (int s = 0; s < 4; ++s) {
            int dcol = dbase + s * 16 + nlan;
            #pragma unroll
            for (int r = 0; r < 8; ++r) {
                int m = t * 16 + r + 8 * half;
                Out[outbase + (size_t)m * DF + dcol] = accO[t][s][r] / lr[t][r];
            }
        }
}

// ---------------------------------------------------------------------------
// Host launcher
// ---------------------------------------------------------------------------
extern "C" void kernel_launch(void* const* d_in, const int* in_sizes, int n_in,
                              void* d_out, int out_size, void* d_ws, size_t ws_size,
                              hipStream_t stream) {
    const float* m_states = (const float*)d_in[0];
    const float* f_k      = (const float*)d_in[1];
    const float* f_v      = (const float*)d_in[2];
    const float* W_q      = (const float*)d_in[3];
    const float* b_q      = (const float*)d_in[4];
    const float* W_k      = (const float*)d_in[5];
    const float* b_k      = (const float*)d_in[6];
    const float* W_v      = (const float*)d_in[7];
    const float* b_v      = (const float*)d_in[8];

    // Workspace layout (~50.5 MB): bf16 Q/K/V + transposed bf16 weights
    char* ws = (char*)d_ws;
    const size_t MB = 1024ull * 1024ull;
    unsigned short* Qb  = (unsigned short*)(ws);
    unsigned short* Kb  = (unsigned short*)(ws + 16 * MB);
    unsigned short* Vb  = (unsigned short*)(ws + 32 * MB);
    unsigned short* WqT = (unsigned short*)(ws + 48 * MB);
    unsigned short* WkT = (unsigned short*)(ws + 49 * MB);
    unsigned short* WvT = (unsigned short*)(ws + 50 * MB);

    wt_convert<<<(DM * DF + 255) / 256, 256, 0, stream>>>(W_q, WqT, DM);
    wt_convert<<<(DF * DF + 255) / 256, 256, 0, stream>>>(W_k, WkT, DF);
    wt_convert<<<(DF * DF + 255) / 256, 256, 0, stream>>>(W_v, WvT, DF);

    dim3 pgrid((NB * SQn) / 128, DF / 64);   // (128, 8)
    proj_gemm<<<pgrid, 256, 0, stream>>>(m_states, WqT, b_q, Qb, DM);
    proj_gemm<<<pgrid, 256, 0, stream>>>(f_k,      WkT, b_k, Kb, DF);
    proj_gemm<<<pgrid, 256, 0, stream>>>(f_v,      WvT, b_v, Vb, DF);

    dim3 agrid(SQn / 32, NB);                // (128, 4)
    attn_fused<<<agrid, 256, 0, stream>>>(Qb, Kb, Vb, (float*)d_out);
}
